// GNN4_71631464562692
// MI455X (gfx1250) — matrix operand order repaired and checked by reading, not compiled, not run on hardware
//
#include <hip/hip_runtime.h>
#include <cstdint>

// ---------------------------------------------------------------------------
// Types for CDNA5 WMMA (gfx1250, wave32)
// ---------------------------------------------------------------------------
typedef __bf16 bf16_t;
typedef bf16_t v16bf __attribute__((ext_vector_type(16)));
typedef bf16_t bfx8  __attribute__((ext_vector_type(8)));
typedef float  v8f   __attribute__((ext_vector_type(8)));

__device__ __forceinline__ bf16_t f2bf(float f) {
  // round-to-nearest-even f32 -> bf16
  unsigned u = __builtin_bit_cast(unsigned, f);
  unsigned r = u + 0x7FFFu + ((u >> 16) & 1u);
  unsigned short h = (unsigned short)(r >> 16);
  return __builtin_bit_cast(bf16_t, h);
}

#define W1T_STRIDE 40  // bf16 elems per row: 80B -> 16B aligned, conflict-free

// ---------------------------------------------------------------------------
// Kernel 1: per-drug fused  dr@W1 + b1 -> relu -> (·w2sum)+b2sum -> softmax
//           -> score-weighted entity gather.  1 block == 1 drug, 8 waves.
//           W1 slabs are double-buffered in LDS; next slab's global loads are
//           issued before the 16 WMMAs so HBM latency overlaps matrix work.
// ---------------------------------------------------------------------------
__global__ __launch_bounds__(256)
void gnn_drug_kernel(const float* __restrict__ drug_table,
                     const float* __restrict__ rela_table,
                     const float* __restrict__ ent_table,
                     const float* __restrict__ W1,
                     const float* __restrict__ b1,
                     const float* __restrict__ W2,
                     const float* __restrict__ b2,
                     const int*   __restrict__ drug_name,
                     const int*   __restrict__ adj_tail,
                     const int*   __restrict__ adj_relation,
                     float*       __restrict__ went)
{
  __shared__ __attribute__((aligned(16))) bf16_t w1t[2][256 * W1T_STRIDE];
  __shared__ float w2sum[256];
  __shared__ float sc[128];
  __shared__ int   tails[128];

  const int b     = blockIdx.x;
  const int tid   = threadIdx.x;
  const int wave  = tid >> 5;     // 0..7
  const int lane  = tid & 31;
  const int ln    = lane & 15;
  const int lhalf = lane >> 4;    // 0/1

  const int dn = drug_name[b];
  const float* __restrict__ drow = drug_table + (size_t)dn * 256;
  const float* __restrict__ W1b  = W1 + (size_t)b * 65536;
  const float* __restrict__ W2b  = W2 + (size_t)b * 65536;

  if (tid < 128) tails[tid] = adj_tail[b * 128 + tid];

  // ---- issue slab-0 global loads now; they complete during phase 0 --------
  float4 pf[8];
#pragma unroll
  for (int i = 0; i < 8; ++i)
    pf[i] = *(const float4*)(W1b + i * 1024 + tid * 4);
  // warm L2 with slab 1 while phase 0 streams W2 (global_prefetch_b8)
  __builtin_prefetch(W1b + 8192 + tid * 32, 0, 3);

  // ---- phase 0a: w2sum[d] = sum_e W2[b][d][e]  (one wave per row) ----------
  for (int dd = wave; dd < 256; dd += 8) {
    const float* row = W2b + dd * 256 + lane * 8;
    float4 a0 = *(const float4*)(row);
    float4 a1 = *(const float4*)(row + 4);
    float s = a0.x + a0.y + a0.z + a0.w + a1.x + a1.y + a1.z + a1.w;
    s += __shfl_xor(s, 1);  s += __shfl_xor(s, 2);  s += __shfl_xor(s, 4);
    s += __shfl_xor(s, 8);  s += __shfl_xor(s, 16);
    if (lane == 0) w2sum[dd] = s;
  }
  // ---- phase 0b: sc[n] = sum_e b2[n][e] ------------------------------------
  for (int n = wave; n < 128; n += 8) {
    const float* row = b2 + n * 256 + lane * 8;
    float4 a0 = *(const float4*)(row);
    float4 a1 = *(const float4*)(row + 4);
    float s = a0.x + a0.y + a0.z + a0.w + a1.x + a1.y + a1.z + a1.w;
    s += __shfl_xor(s, 1);  s += __shfl_xor(s, 2);  s += __shfl_xor(s, 4);
    s += __shfl_xor(s, 8);  s += __shfl_xor(s, 16);
    if (lane == 0) sc[n] = s;
  }

  // ---- phase A: h = relu(dr @ W1 + b1), accumulated in registers -----------
  const int rt = wave;                 // row tile: rows 16*rt .. 16*rt+15
  const int M  = 16 * rt + ln;         // A-matrix row owned by this lane
  const int rel = adj_relation[b * 128 + M];
  const float* __restrict__ rrow = rela_table + (size_t)rel * 256;

  v8f acc[16];
  const v8f vzero = {0.f, 0.f, 0.f, 0.f, 0.f, 0.f, 0.f, 0.f};
#pragma unroll
  for (int ct = 0; ct < 16; ++ct) acc[ct] = vzero;

  for (int ks = 0; ks < 8; ++ks) {
    // store the already-loaded slab ks into ping-pong buffer (transposed bf16)
    bf16_t* __restrict__ buf = &w1t[ks & 1][0];
#pragma unroll
    for (int i = 0; i < 8; ++i) {
      const int fi = i * 1024 + tid * 4;       // fi = kk*256 + n
      const int kk = fi >> 8;
      const int n  = fi & 255;
      buf[(n + 0) * W1T_STRIDE + kk] = f2bf(pf[i].x);
      buf[(n + 1) * W1T_STRIDE + kk] = f2bf(pf[i].y);
      buf[(n + 2) * W1T_STRIDE + kk] = f2bf(pf[i].z);
      buf[(n + 3) * W1T_STRIDE + kk] = f2bf(pf[i].w);
    }
    __syncthreads();   // slab ks visible; also fences phase 0 / prior reads

    // issue global loads for slab ks+1 -- overlaps with the 16 WMMAs below
    if (ks < 7) {
      const float* Wk = W1b + (ks + 1) * 8192;
#pragma unroll
      for (int i = 0; i < 8; ++i)
        pf[i] = *(const float4*)(Wk + i * 1024 + tid * 4);
    }

    // A fragment: dr[M][k0 + off..off+7] and [k0 + 16+off..], off = 8*lhalf
    const int k0  = ks * 32;
    const int off = lhalf * 8;
    v16bf afrag;
    {
      const float* rp = rrow + k0 + off;
      const float* dp = drow + k0 + off;
      float4 r0 = *(const float4*)(rp);      float4 d0 = *(const float4*)(dp);
      float4 r1 = *(const float4*)(rp + 4);  float4 d1 = *(const float4*)(dp + 4);
      float4 r2 = *(const float4*)(rp + 16); float4 d2 = *(const float4*)(dp + 16);
      float4 r3 = *(const float4*)(rp + 20); float4 d3 = *(const float4*)(dp + 20);
      afrag[0]  = f2bf(r0.x * d0.x); afrag[1]  = f2bf(r0.y * d0.y);
      afrag[2]  = f2bf(r0.z * d0.z); afrag[3]  = f2bf(r0.w * d0.w);
      afrag[4]  = f2bf(r1.x * d1.x); afrag[5]  = f2bf(r1.y * d1.y);
      afrag[6]  = f2bf(r1.z * d1.z); afrag[7]  = f2bf(r1.w * d1.w);
      afrag[8]  = f2bf(r2.x * d2.x); afrag[9]  = f2bf(r2.y * d2.y);
      afrag[10] = f2bf(r2.z * d2.z); afrag[11] = f2bf(r2.w * d2.w);
      afrag[12] = f2bf(r3.x * d3.x); afrag[13] = f2bf(r3.y * d3.y);
      afrag[14] = f2bf(r3.z * d3.z); afrag[15] = f2bf(r3.w * d3.w);
    }

#pragma unroll
    for (int ct = 0; ct < 16; ++ct) {
      const int nb = (16 * ct + ln) * W1T_STRIDE + lhalf * 8;
      bfx8 blo = *(const bfx8*)&buf[nb];
      bfx8 bhi = *(const bfx8*)&buf[nb + 16];
      v16bf bfrag = __builtin_shufflevector(blo, bhi,
          0, 1, 2, 3, 4, 5, 6, 7, 8, 9, 10, 11, 12, 13, 14, 15);
      acc[ct] = __builtin_amdgcn_wmma_f32_16x16x32_bf16(
          /*neg_a=*/false, afrag, /*neg_b=*/false, bfrag,
          /*c_mod=*/(short)0, acc[ct], /*reuse_a=*/false, /*reuse_b=*/false);
    }
  }

  // ---- phase B: logits  sc[row] += sum_col relu(acc + b1) * w2sum[col] -----
  {
    float part[8];
#pragma unroll
    for (int i = 0; i < 8; ++i) part[i] = 0.f;
#pragma unroll
    for (int ct = 0; ct < 16; ++ct) {
      const int col = 16 * ct + ln;
      const float wsc = w2sum[col];
      const float* b1p = b1 + (16 * rt + 8 * lhalf) * 256 + col;
#pragma unroll
      for (int i = 0; i < 8; ++i) {
        float v = acc[ct][i] + b1p[i * 256];
        v = fmaxf(v, 0.f);
        part[i] = fmaf(v, wsc, part[i]);
      }
    }
#pragma unroll
    for (int i = 0; i < 8; ++i) {   // reduce across the 16 lanes of each half
      float p = part[i];
      p += __shfl_xor(p, 1); p += __shfl_xor(p, 2);
      p += __shfl_xor(p, 4); p += __shfl_xor(p, 8);
      if (ln == 0) sc[16 * rt + 8 * lhalf + i] += p;
    }
  }
  __syncthreads();

  // ---- phase C: softmax over 128 logits (wave 0) ---------------------------
  if (wave == 0) {
    float v0 = sc[lane], v1 = sc[lane + 32], v2 = sc[lane + 64], v3 = sc[lane + 96];
    float m = fmaxf(fmaxf(v0, v1), fmaxf(v2, v3));
#pragma unroll
    for (int o = 16; o > 0; o >>= 1) m = fmaxf(m, __shfl_xor(m, o));
    float e0 = __expf(v0 - m), e1 = __expf(v1 - m);
    float e2 = __expf(v2 - m), e3 = __expf(v3 - m);
    float s = e0 + e1 + e2 + e3;
#pragma unroll
    for (int o = 16; o > 0; o >>= 1) s += __shfl_xor(s, o);
    const float inv = 1.f / s;
    sc[lane] = e0 * inv; sc[lane + 32] = e1 * inv;
    sc[lane + 64] = e2 * inv; sc[lane + 96] = e3 * inv;
  }
  __syncthreads();

  // ---- phase D: w_ent[k] = sum_n score[n] * ent_table[tail[n]][k] ----------
  {
    float a = 0.f;
    const int k = tid;
    for (int n = 0; n < 128; ++n)
      a = fmaf(sc[n], ent_table[(size_t)tails[n] * 256 + k], a);
    went[(size_t)b * 256 + k] = a;
  }
}

// ---------------------------------------------------------------------------
// Kernel 2: y = relu([w_ent, d] @ lin_w^T + lin_b)   (lin_w L2-resident)
// ---------------------------------------------------------------------------
__global__ __launch_bounds__(256)
void gnn_lin_kernel(const float* __restrict__ went,
                    const float* __restrict__ drug_table,
                    const int*   __restrict__ drug_name,
                    const float* __restrict__ lin_w,
                    const float* __restrict__ lin_b,
                    float*       __restrict__ y)
{
  const int b = blockIdx.x;
  const int j = threadIdx.x;
  __shared__ __attribute__((aligned(16))) float x[512];

  const int dn = drug_name[b];
  x[j]       = went[(size_t)b * 256 + j];
  x[256 + j] = drug_table[(size_t)dn * 256 + j];
  __syncthreads();

  const float* wr = lin_w + (size_t)j * 512;
  float a = lin_b[j];
#pragma unroll 4
  for (int k = 0; k < 512; k += 4) {
    float4 wv = *(const float4*)(wr + k);
    float4 xv = *(const float4*)(x + k);
    a = fmaf(wv.x, xv.x, a); a = fmaf(wv.y, xv.y, a);
    a = fmaf(wv.z, xv.z, a); a = fmaf(wv.w, xv.w, a);
  }
  y[(size_t)b * 256 + j] = fmaxf(a, 0.f);
}

// ---------------------------------------------------------------------------
// Kernel 3: BatchNorm1d (training mode, biased variance) over batch axis.
// One block per feature column.
// ---------------------------------------------------------------------------
__global__ __launch_bounds__(256)
void gnn_bn_kernel(const float* __restrict__ y,
                   const float* __restrict__ bn_gamma,
                   const float* __restrict__ bn_beta,
                   float*       __restrict__ out)
{
  const int j = blockIdx.x;
  const int t = threadIdx.x;
  __shared__ float rs[256];
  __shared__ float rs2[256];

  float s = 0.f, s2 = 0.f;
  for (int bb = t; bb < 572; bb += 256) {
    const float v = y[(size_t)bb * 256 + j];
    s += v;
    s2 += v * v;
  }
  rs[t] = s; rs2[t] = s2;
  __syncthreads();
  for (int o = 128; o > 0; o >>= 1) {
    if (t < o) { rs[t] += rs[t + o]; rs2[t] += rs2[t + o]; }
    __syncthreads();
  }
  const float mean = rs[0] * (1.f / 572.f);
  const float var  = rs2[0] * (1.f / 572.f) - mean * mean;
  const float inv  = rsqrtf(var + 1e-5f);
  const float g = bn_gamma[j], be = bn_beta[j];
  for (int bb = t; bb < 572; bb += 256)
    out[(size_t)bb * 256 + j] = g * (y[(size_t)bb * 256 + j] - mean) * inv + be;
}

// ---------------------------------------------------------------------------
// Launcher
// ---------------------------------------------------------------------------
extern "C" void kernel_launch(void* const* d_in, const int* in_sizes, int n_in,
                              void* d_out, int out_size, void* d_ws, size_t ws_size,
                              hipStream_t stream) {
  const float* drug_table   = (const float*)d_in[0];
  const float* rela_table   = (const float*)d_in[1];
  const float* ent_table    = (const float*)d_in[2];
  const float* W1           = (const float*)d_in[3];
  const float* b1           = (const float*)d_in[4];
  const float* W2           = (const float*)d_in[5];
  const float* b2           = (const float*)d_in[6];
  const float* lin_w        = (const float*)d_in[7];
  const float* lin_b        = (const float*)d_in[8];
  const float* bn_gamma     = (const float*)d_in[9];
  const float* bn_beta      = (const float*)d_in[10];
  const int*   drug_name    = (const int*)d_in[11];
  const int*   adj_tail     = (const int*)d_in[12];
  const int*   adj_relation = (const int*)d_in[13];

  float* went = (float*)d_ws;                 // 572*256 f32
  float* y    = went + 572 * 256;             // 572*256 f32
  float* out  = (float*)d_out;

  gnn_drug_kernel<<<572, 256, 0, stream>>>(drug_table, rela_table, ent_table,
                                           W1, b1, W2, b2,
                                           drug_name, adj_tail, adj_relation,
                                           went);
  gnn_lin_kernel<<<572, 256, 0, stream>>>(went, drug_table, drug_name,
                                          lin_w, lin_b, y);
  gnn_bn_kernel<<<256, 256, 0, stream>>>(y, bn_gamma, bn_beta, out);
}
